// P4GRU_41858751266909
// MI455X (gfx1250) — matrix operand order repaired and checked by reading, not compiled, not run on hardware
//
#include <hip/hip_runtime.h>
#include <math.h>

typedef __attribute__((ext_vector_type(16))) __bf16 v16bf;
typedef __attribute__((ext_vector_type(8)))  __bf16 v8bf;
typedef __attribute__((ext_vector_type(8)))  float  v8f;

__device__ __forceinline__ float apply_act(float v, int act) {
  if (act == 1) return 1.f / (1.f + __expf(-v));                 // sigmoid
  if (act == 2) return tanhf(v);                                 // tanh
  if (act == 3) return fmaxf(v, 0.f);                            // relu
  if (act == 4) {                                                // gelu (tanh approx)
    float u = 0.7978845608028654f * (v + 0.044715f * v * v * v);
    return 0.5f * v * (1.f + tanhf(u));
  }
  return v;
}

// ---------------------------------------------------------------------------
// Guard-free WMMA GEMM on pre-packed bf16 operands.
//   A  : bf16 [>=ceil64(M) rows][lda], logical K padded to Kp (mult of 32),
//        zero-filled padding.
//   Bt : bf16 [>=ceil32(N) rows][ldb], row n = column n of B (16 contiguous
//        K values per fragment -> pure global_load_b128, no LDS needed).
//   C  : f32 [M][ldc] (+bias)(+act), masked store only in epilogue.
//   Block 64x32 tile, 128 thr = 4 waves, wave w owns rows w*16..w*16+15 and
//   both 16-wide N tiles (A fragment reused across 2 WMMAs).
//   The K loop contains NO branches/divergence: EXEC all-ones for v_wmma.
// ---------------------------------------------------------------------------
__global__ __launch_bounds__(128) void k_gemm(
    const __bf16* __restrict__ A, const __bf16* __restrict__ Bt,
    float* __restrict__ Cm, const float* __restrict__ bias,
    int M, int N, int Kp, int lda, int ldb, int ldc,
    long long sA, long long sB, long long sC, int biasMode, int act)
{
  const int lane = threadIdx.x & 31;
  const int wave = threadIdx.x >> 5;
  const int mbase = blockIdx.y * 64;
  const int nbase = blockIdx.x * 32;
  A  += (long long)blockIdx.z * sA;
  Bt += (long long)blockIdx.z * sB;
  Cm += (long long)blockIdx.z * sC;

  const int hi  = (lane >> 4) & 1;
  const int l15 = lane & 15;
  const long rowA  = mbase + wave * 16 + l15;
  const long colB0 = nbase + l15;

  // A fragment: elems j = K kb + j + 16*hi (contiguous in A row)
  const __bf16* ap  = A  + rowA * lda + hi * 16;
  // B fragment: elems j = K kb + j + 16*hi (contiguous in Bt row)
  const __bf16* bp0 = Bt + colB0 * ldb + hi * 16;
  const __bf16* bp1 = bp0 + (long)16 * ldb;

  v8f acc0 = {0.f,0.f,0.f,0.f,0.f,0.f,0.f,0.f};
  v8f acc1 = {0.f,0.f,0.f,0.f,0.f,0.f,0.f,0.f};

  for (int kb = 0; kb < Kp; kb += 32) {
    v8bf alo = *(const v8bf*)(ap + kb);
    v8bf ahi = *(const v8bf*)(ap + kb + 8);
    v16bf af = __builtin_shufflevector(alo, ahi,
        0,1,2,3,4,5,6,7, 8,9,10,11,12,13,14,15);
    v16bf b0 = *(const v16bf*)(bp0 + kb);
    v16bf b1 = *(const v16bf*)(bp1 + kb);
    if (kb + 32 < Kp)
      __builtin_prefetch(ap + kb + 32, 0, 0);   // global_prefetch_b8, uniform
    acc0 = __builtin_amdgcn_wmma_f32_16x16x32_bf16(
        false, af, false, b0, (short)0, acc0, false, false);
    acc1 = __builtin_amdgcn_wmma_f32_16x16x32_bf16(
        false, af, false, b1, (short)0, acc1, false, false);
  }

#pragma unroll
  for (int r = 0; r < 8; ++r) {
    int row = mbase + wave * 16 + r + hi * 8;
    if (row < M) {
      int col0 = nbase + l15;
      int col1 = col0 + 16;
      float v0 = acc0[r], v1 = acc1[r];
      if (biasMode == 1) { v0 += bias[row]; v1 += bias[row]; }
      else if (biasMode == 2) {
        if (col0 < N) v0 += bias[col0];
        if (col1 < N) v1 += bias[col1];
      }
      if (col0 < N) Cm[(long)row * ldc + col0] = apply_act(v0, act);
      if (col1 < N) Cm[(long)row * ldc + col1] = apply_act(v1, act);
    }
  }
}

// ---- packing helpers (one-shot, branchy loads acceptable here) ------------
__global__ void k_packA(const float* __restrict__ src, __bf16* __restrict__ dst,
                        int M, int K, int ld, int Kp, int rows)
{
  long i = (long)blockIdx.x * blockDim.x + threadIdx.x;
  if (i >= (long)rows * Kp) return;
  int r = i / Kp, k = i % Kp;
  float v = (r < M && k < K) ? src[(long)r * ld + k] : 0.f;
  dst[i] = (__bf16)v;
}
__global__ void k_packBt(const float* __restrict__ src, __bf16* __restrict__ dst,
                         int N, int K, int ld, int Kp, int trans, int rows)
{
  long i = (long)blockIdx.x * blockDim.x + threadIdx.x;
  if (i >= (long)rows * Kp) return;
  int n = i / Kp, k = i % Kp;
  float v = 0.f;
  if (n < N && k < K) v = trans ? src[(long)n * ld + k] : src[(long)k * ld + n];
  dst[i] = (__bf16)v;
}
__global__ void k_castbf(__bf16* __restrict__ dst, const float* __restrict__ src, long n)
{
  long i = (long)blockIdx.x * blockDim.x + threadIdx.x;
  if (i < n) dst[i] = (__bf16)src[i];
}

// ---------------------------------------------------------------------------
__global__ void k_fps(const float* __restrict__ xyz, int* __restrict__ idx,
                      int N, int npoint, long long sXyz, long long sIdx)
{
  xyz += blockIdx.x * sXyz;
  idx += blockIdx.x * sIdx;
  __shared__ float dist[2048];
  __shared__ float rmax[256];
  __shared__ int   rarg[256];
  __shared__ int   sfar;
  int tid = threadIdx.x;
  for (int i = tid; i < N; i += 256) dist[i] = 1e10f;
  if (tid == 0) { sfar = 0; idx[0] = 0; }
  __syncthreads();
  for (int it = 1; it < npoint; ++it) {
    int far = sfar;
    float cx = xyz[far*3], cy = xyz[far*3+1], cz = xyz[far*3+2];
    float best = -1.f; int barg = 0;
    for (int i = tid; i < N; i += 256) {
      float dx = xyz[i*3] - cx, dy = xyz[i*3+1] - cy, dz = xyz[i*3+2] - cz;
      float nd = fminf(dist[i], dx*dx + dy*dy + dz*dz);
      dist[i] = nd;
      if (nd > best) { best = nd; barg = i; }
    }
    rmax[tid] = best; rarg[tid] = barg;
    __syncthreads();
    for (int o = 128; o > 0; o >>= 1) {
      if (tid < o && rmax[tid + o] > rmax[tid]) { rmax[tid] = rmax[tid+o]; rarg[tid] = rarg[tid+o]; }
      __syncthreads();
    }
    if (tid == 0) { sfar = rarg[0]; idx[it] = rarg[0]; }
    __syncthreads();
  }
}

__global__ void k_gather3(const float* __restrict__ src, const int* __restrict__ idx,
                          float* __restrict__ out, int M,
                          long long sSrc, long long sIdx, long long sOut)
{
  int i = blockIdx.x * blockDim.x + threadIdx.x;
  int b = i / M, m = i % M;
  if (b >= 2) return;
  int j = idx[b * sIdx + m];
  const float* s = src + b * sSrc + (long)j * 3;
  float* o = out + b * sOut + (long)m * 3;
  o[0] = s[0]; o[1] = s[1]; o[2] = s[2];
}

__global__ void k_ball_query(const float* __restrict__ nxyz, const float* __restrict__ xyz,
                             int* __restrict__ idx, int M, int N, int ns, float r2,
                             long long sN, long long sX, long long sI)
{
  int m = blockIdx.x * blockDim.x + threadIdx.x;
  int b = blockIdx.y;
  if (m >= M) return;
  const float* p = nxyz + b * sN + (long)m * 3;
  const float* q = xyz + b * sX;
  int* o = idx + b * sI + (long)m * ns;
  float px = p[0], py = p[1], pz = p[2];
  int cnt = 0, first = 0;
  for (int j = 0; j < N && cnt < ns; ++j) {
    float dx = q[j*3] - px, dy = q[j*3+1] - py, dz = q[j*3+2] - pz;
    if (dx*dx + dy*dy + dz*dz <= r2) { if (cnt == 0) first = j; o[cnt++] = j; }
  }
  int pad = (cnt > 0) ? first : 0;
  for (; cnt < ns; ++cnt) o[cnt] = pad;
}

// corr row (m*ns+s) = [ S2[:,idx] ; X1[:,m] ; P2[idx]-P1[m] ; 0-pad ] as bf16 Bt
__global__ void k_build_corr(const float* __restrict__ S2, const float* __restrict__ X1,
                             const float* __restrict__ P1, const float* __restrict__ P2,
                             const int* __restrict__ idx, __bf16* __restrict__ corr,
                             int C2, int C1, int Mn, int ns, int N2, int Cp)
{
  int i = blockIdx.x * blockDim.x + threadIdx.x;
  if (i >= Mn * ns) return;
  int m = i / ns;
  int j = idx[i];
  __bf16* row = corr + (long)i * Cp;
  int c = 0;
  for (; c < C2; ++c) row[c] = (__bf16)S2[(long)c * N2 + j];
  for (int cc = 0; cc < C1; ++cc, ++c) row[c] = (__bf16)X1[(long)cc * Mn + m];
  row[c++] = (__bf16)(P2[j*3]   - P1[m*3]);
  row[c++] = (__bf16)(P2[j*3+1] - P1[m*3+1]);
  row[c++] = (__bf16)(P2[j*3+2] - P1[m*3+2]);
  for (; c < Cp; ++c) row[c] = (__bf16)0.f;
}

__global__ void k_maxpool(const float* __restrict__ in, const float* __restrict__ bias,
                          float* __restrict__ out, int O, int Mn, int ns, int act)
{
  int i = blockIdx.x * blockDim.x + threadIdx.x;
  if (i >= O * Mn) return;
  int o = i / Mn, m = i % Mn;
  const float* p = in + (long)o * Mn * ns + (long)m * ns;
  float v = -1e30f;
  for (int s = 0; s < ns; ++s) v = fmaxf(v, p[s]);
  v += bias[o];
  out[i] = apply_act(v, act);
}

__global__ void k_group_max(const float* __restrict__ feat, const int* __restrict__ idx,
                            float* __restrict__ out, int C, int M, int N, int ns)
{
  int b = blockIdx.y;
  int i = blockIdx.x * blockDim.x + threadIdx.x;
  if (i >= C * M) return;
  int c = i / M, m = i % M;
  const float* f = feat + (long)b * C * N + (long)c * N;
  const int* id = idx + (long)b * M * ns + (long)m * ns;
  float v = -1e30f;
  for (int s = 0; s < ns; ++s) v = fmaxf(v, f[id[s]]);
  out[(long)b * C * M + i] = v;
}

__global__ void k_fill(float* __restrict__ p, float v, long n) {
  long i = (long)blockIdx.x * blockDim.x + threadIdx.x;
  if (i < n) p[i] = v;
}
__global__ void k_copy(float* __restrict__ d, const float* __restrict__ s, long n) {
  long i = (long)blockIdx.x * blockDim.x + threadIdx.x;
  if (i < n) d[i] = s[i];
}
__global__ void k_add(float* __restrict__ d, const float* __restrict__ s, long n) {
  long i = (long)blockIdx.x * blockDim.x + threadIdx.x;
  if (i < n) d[i] += s[i];
}
__global__ void k_copy_b(float* __restrict__ d, const float* __restrict__ s,
                         long per, long sd, long ss, int batch) {
  long i = (long)blockIdx.x * blockDim.x + threadIdx.x;
  if (i >= per * batch) return;
  long b = i / per, j = i % per;
  d[b * sd + j] = s[b * ss + j];
}
__global__ void k_gru_combine(float* __restrict__ S, const float* __restrict__ Z,
                              const float* __restrict__ So, const float* __restrict__ Sn, long n) {
  long i = (long)blockIdx.x * blockDim.x + threadIdx.x;
  if (i < n) S[i] = Z[i] * So[i] + (1.f - Z[i]) * Sn[i];
}

// dst Bt[n][c] = bf16( R[c][n] * So[c][n] )        (grid.y = batch)
__global__ void k_mul_pack(__bf16* __restrict__ dst, const float* __restrict__ a,
                           const float* __restrict__ b, int C, int Mn,
                           long sd, long sa)
{
  int bb = blockIdx.y;
  int i = blockIdx.x * blockDim.x + threadIdx.x;
  if (i >= C * Mn) return;
  int n = i / C, c = i % C;
  long so = bb * sa + (long)c * Mn + n;
  dst[bb * sd + i] = (__bf16)(a[so] * b[so]);
}

// fc input Bt[n][k]: k<C1 -> X1[k][n], else R[k-C1][n]*So[k-C1][n]   (grid.y=b)
__global__ void k_fcin_pack(__bf16* __restrict__ dst, const float* __restrict__ X1,
                            const float* __restrict__ R, const float* __restrict__ So,
                            int C1, int C2, int Mn, long sd, long sx, long srs)
{
  int bb = blockIdx.y;
  int i = blockIdx.x * blockDim.x + threadIdx.x;
  int Ct = C1 + C2;
  if (i >= Ct * Mn) return;
  int n = i / Ct, k = i % Ct;
  float v;
  if (k < C1) v = X1[bb * sx + (long)k * Mn + n];
  else {
    long so = bb * srs + (long)(k - C1) * Mn + n;
    v = R[so] * So[so];
  }
  dst[bb * sd + i] = (__bf16)v;
}

__global__ void k_layernorm(const float* __restrict__ x, const float* __restrict__ g,
                            const float* __restrict__ b, __bf16* __restrict__ y, int dim)
{
  long row = blockIdx.x;
  x += row * dim; y += row * dim;
  __shared__ float red[256];
  int tid = threadIdx.x;
  float s = 0.f;
  for (int i = tid; i < dim; i += 256) s += x[i];
  red[tid] = s; __syncthreads();
  for (int o = 128; o > 0; o >>= 1) { if (tid < o) red[tid] += red[tid + o]; __syncthreads(); }
  float mean = red[0] / dim; __syncthreads();
  float v = 0.f;
  for (int i = tid; i < dim; i += 256) { float d = x[i] - mean; v += d * d; }
  red[tid] = v; __syncthreads();
  for (int o = 128; o > 0; o >>= 1) { if (tid < o) red[tid] += red[tid + o]; __syncthreads(); }
  float rstd = rsqrtf(red[0] / dim + 1e-5f);
  for (int i = tid; i < dim; i += 256)
    y[i] = (__bf16)((x[i] - mean) * rstd * g[i] + b[i]);
}

__global__ void k_softmax(const float* __restrict__ s, __bf16* __restrict__ o,
                          int n, float scale)
{
  s += (long)blockIdx.x * n;
  o += (long)blockIdx.x * n;
  __shared__ float red[256];
  int tid = threadIdx.x;
  float mx = -1e30f;
  for (int i = tid; i < n; i += 256) mx = fmaxf(mx, s[i] * scale);
  red[tid] = mx; __syncthreads();
  for (int off = 128; off > 0; off >>= 1) { if (tid < off) red[tid] = fmaxf(red[tid], red[tid + off]); __syncthreads(); }
  mx = red[0]; __syncthreads();
  float sum = 0.f;
  for (int i = tid; i < n; i += 256) sum += __expf(s[i] * scale - mx);
  red[tid] = sum; __syncthreads();
  for (int off = 128; off > 0; off >>= 1) { if (tid < off) red[tid] += red[tid + off]; __syncthreads(); }
  float inv = 1.f / red[0];
  for (int i = tid; i < n; i += 256) o[i] = (__bf16)(__expf(s[i] * scale - mx) * inv);
}

__global__ void k_relu_maxtok(const float* __restrict__ x, float* __restrict__ out, int M, int C)
{
  int b = blockIdx.y;
  int c = blockIdx.x * blockDim.x + threadIdx.x;
  if (c >= C) return;
  x += (long)b * M * C;
  float v = -1e30f;
  for (int m = 0; m < M; ++m) v = fmaxf(v, x[(long)m * C + c]);
  out[(long)b * C + c] = fmaxf(v, 0.f);
}

// ---------------------------------------------------------------------------
static void gemm(hipStream_t st, const __bf16* A, const __bf16* Bt, float* C,
                 const float* bias, int M, int N, int Kp, int lda, int ldb, int ldc,
                 long long sA, long long sB, long long sC, int batch,
                 int biasMode, int act)
{
  dim3 g((N + 31) / 32, (M + 63) / 64, batch);
  k_gemm<<<g, dim3(128), 0, st>>>(A, Bt, C, bias, M, N, Kp, lda, ldb, ldc,
                                  sA, sB, sC, biasMode, act);
}

extern "C" void kernel_launch(void* const* d_in, const int* in_sizes, int n_in,
                              void* d_out, int out_size, void* d_ws, size_t ws_size,
                              hipStream_t stream)
{
  (void)in_sizes; (void)n_in; (void)out_size; (void)ws_size;
  const float* points = (const float*)d_in[0];
  const float* zW0 = (const float*)d_in[1];  const float* zb0 = (const float*)d_in[2];
  const float* rW0 = (const float*)d_in[3];  const float* rb0 = (const float*)d_in[4];
  const float* sW0 = (const float*)d_in[5];  const float* sb0 = (const float*)d_in[6];
  const float* fW0 = (const float*)d_in[7];  const float* fb0 = (const float*)d_in[8];
  const float* zW1 = (const float*)d_in[9];  const float* zb1 = (const float*)d_in[10];
  const float* rW1 = (const float*)d_in[11]; const float* rb1 = (const float*)d_in[12];
  const float* sW1 = (const float*)d_in[13]; const float* sb1 = (const float*)d_in[14];
  const float* fW1 = (const float*)d_in[15]; const float* fb1 = (const float*)d_in[16];
  const float* posW = (const float*)d_in[17]; const float* posb = (const float*)d_in[18];
  const float* hlng = (const float*)d_in[41]; const float* hlnb = (const float*)d_in[42];
  const float* l1W = (const float*)d_in[43];  const float* l1b = (const float*)d_in[44];
  const float* l2W = (const float*)d_in[45];  const float* l2b = (const float*)d_in[46];

  float* Wp = (float*)d_ws;
  size_t off = 0;
  auto alloc   = [&](size_t n) { float* p = Wp + off; off += (n + 63) & ~(size_t)63; return p; };
  auto allocbf = [&](size_t n) { return (__bf16*)alloc((n + 1) / 2); };

  // f32 buffers
  float* gbuf    = alloc(4194304);   // z/r/s GEMM out: max(128*32768, 256*6144)
  float* scores  = alloc(4194304);   // 2*8*512*512
  float* qkv     = alloc(786432);    // 2*512*768
  float* ffb     = alloc(524288);    // 2*512*512
  float* xb      = alloc(262144);    // 2*512*256
  float* hb      = alloc(262144);
  float* ob      = alloc(262144);
  float* featsAll= alloc(262144);
  float* possAll = alloc(3072);
  float* S0  = alloc(131072);
  float* P0  = alloc(3072);
  float* xyz0= alloc(3072);
  float* Z0  = alloc(131072); float* R0 = alloc(131072);
  float* So0 = alloc(131072); float* Sn0 = alloc(131072);
  float* xyz1 = alloc(768);
  float* feat1= alloc(32768);
  float* S1  = alloc(65536);
  float* P1s = alloc(768);
  float* Z1  = alloc(65536); float* R1 = alloc(65536);
  float* So1 = alloc(65536); float* Sn1 = alloc(65536);
  float* embmax = alloc(512); float* l1o = alloc(1024);
  int* idx0 = (int*)alloc(1024);
  int* bq0  = (int*)alloc(65536);
  int* idx1 = (int*)alloc(256);
  int* bqf  = (int*)alloc(4096);
  int* bq1  = (int*)alloc(12288);

  // bf16 buffers (packed operands)
  __bf16* corrbf  = allocbf(5242880);  // max(32768*160, 6144*416)
  __bf16* RS0bf   = allocbf(131072);   // 2 x [512][128]
  __bf16* fcin1bf = allocbf(98304);    // 2 x [128][384]
  __bf16* hbf     = allocbf(262144);   // layernorm out, 2 x [512][256]
  __bf16* qkvbf   = allocbf(786432);
  __bf16* vtbf    = allocbf(262144);   // 2*8 x [32][512]
  __bf16* scorebf = allocbf(4194304);
  __bf16* obbf    = allocbf(262144);
  __bf16* ffbbf   = allocbf(524288);
  __bf16* posAbf  = allocbf(32768);    // 2 x [512][32]
  __bf16* hlnbf   = allocbf(16384);    // [64][256] (rows 2..63 zero)
  __bf16* l1obf   = allocbf(32768);    // [64][512] (rows 2..63 zero)
  // packed weights
  __bf16* zW0p = allocbf(20480);  __bf16* rW0p = allocbf(20480);
  __bf16* sW0p = allocbf(20480);  __bf16* fW0p = allocbf(16384);
  __bf16* zW1p = allocbf(106496); __bf16* rW1p = allocbf(106496);
  __bf16* sW1p = allocbf(73728);  __bf16* fW1p = allocbf(98304);
  __bf16* posWp = allocbf(8192);
  __bf16* qkvWp[2] = { allocbf(196608), allocbf(196608) };
  __bf16* outWp[2] = { allocbf(65536),  allocbf(65536)  };
  __bf16* ff1Wp[2] = { allocbf(131072), allocbf(131072) };
  __bf16* ff2Wp[2] = { allocbf(131072), allocbf(131072) };
  __bf16* l1Wp = allocbf(131072); __bf16* l2Wp = allocbf(262144);

  const int THR = 256;
  auto blk = [](long n) { return dim3((unsigned)((n + 255) / 256)); };
  auto packA = [&](const float* s, __bf16* d, int M, int K, int ld, int Kp, int rows) {
    k_packA<<<blk((long)rows * Kp), THR, 0, stream>>>(s, d, M, K, ld, Kp, rows);
  };
  auto packBt = [&](const float* s, __bf16* d, int N, int K, int ld, int Kp, int trans, int rows) {
    k_packBt<<<blk((long)rows * Kp), THR, 0, stream>>>(s, d, N, K, ld, Kp, trans, rows);
  };

  // ---- pack all weights once ----
  packA(zW0, zW0p, 128, 131, 131, 160, 128);
  packA(rW0, rW0p, 128, 131, 131, 160, 128);
  packA(sW0, sW0p, 128, 131, 131, 160, 128);
  packA(fW0, fW0p, 128, 128, 128, 128, 128);
  packA(zW1, zW1p, 256, 387, 387, 416, 256);
  packA(rW1, rW1p, 256, 387, 387, 416, 256);
  packA(sW1, sW1p, 256, 259, 259, 288, 256);
  packA(fW1, fW1p, 256, 384, 384, 384, 256);
  packBt(posW, posWp, 256, 3, 3, 32, 1, 256);
  for (int l = 0; l < 2; ++l) {
    packBt((const float*)d_in[21 + 11*l], qkvWp[l], 768, 256, 256, 256, 1, 768);
    packBt((const float*)d_in[22 + 11*l], outWp[l], 256, 256, 256, 256, 1, 256);
    packBt((const float*)d_in[26 + 11*l], ff1Wp[l], 512, 256, 256, 256, 1, 512);
    packBt((const float*)d_in[28 + 11*l], ff2Wp[l], 256, 512, 512, 512, 1, 256);
  }
  packBt(l1W, l1Wp, 512, 256, 256, 256, 1, 512);
  packBt(l2W, l2Wp, 512, 512, 512, 512, 1, 512);
  k_fill<<<blk(8192), THR, 0, stream>>>((float*)hlnbf, 0.f, 8192);    // zero pad rows
  k_fill<<<blk(16384), THR, 0, stream>>>((float*)l1obf, 0.f, 16384);

  for (int t = 0; t < 4; ++t) {
    const float* pts_t = points + (long)t * 2048 * 3;
    // ---- level 0 ----
    k_fps<<<dim3(2), dim3(256), 0, stream>>>(pts_t, idx0, 2048, 512, (long long)4*2048*3, 512);
    k_gather3<<<blk(2*512), THR, 0, stream>>>(pts_t, idx0, xyz0, 512, (long long)4*2048*3, 512, 512*3);
    if (t == 0) {
      k_fill<<<blk(131072), THR, 0, stream>>>(S0, 0.f, 131072);
      k_copy<<<blk(3072), THR, 0, stream>>>(P0, xyz0, 3072);
    }
    k_ball_query<<<dim3(8, 2), dim3(64), 0, stream>>>(xyz0, P0, bq0, 512, 512, 64,
                                                      0.12f*0.12f, 512*3, 512*3, 512*64);
    for (int b = 0; b < 2; ++b) {
      const int MNS = 512 * 64;
      k_build_corr<<<blk(MNS), THR, 0, stream>>>(S0 + b*65536, nullptr, xyz0 + b*1536,
                                                 P0 + b*1536, bq0 + b*MNS, corrbf,
                                                 128, 0, 512, 64, 512, 160);
      gemm(stream, zW0p, corrbf, gbuf, nullptr, 128, MNS, 160, 160, 160, MNS, 0,0,0,1, 0,0);
      k_maxpool<<<blk(128*512), THR, 0, stream>>>(gbuf, zb0, Z0 + b*65536, 128, 512, 64, 1);
      gemm(stream, rW0p, corrbf, gbuf, nullptr, 128, MNS, 160, 160, 160, MNS, 0,0,0,1, 0,0);
      k_maxpool<<<blk(128*512), THR, 0, stream>>>(gbuf, rb0, R0 + b*65536, 128, 512, 64, 1);
      gemm(stream, sW0p, corrbf, gbuf, nullptr, 128, MNS, 160, 160, 160, MNS, 0,0,0,1, 0,0);
      k_maxpool<<<blk(128*512), THR, 0, stream>>>(gbuf, sb0, So0 + b*65536, 128, 512, 64, 0);
    }
    k_mul_pack<<<dim3((512*128 + 255)/256, 2), THR, 0, stream>>>(RS0bf, R0, So0, 128, 512,
                                                                 65536, 65536);
    for (int b = 0; b < 2; ++b)
      gemm(stream, fW0p, RS0bf + b*65536, Sn0 + b*65536, fb0, 128, 512, 128,
           128, 128, 512, 0,0,0,1, 1,2);
    k_gru_combine<<<blk(131072), THR, 0, stream>>>(S0, Z0, So0, Sn0, 131072);
    k_copy<<<blk(3072), THR, 0, stream>>>(P0, xyz0, 3072);
    // ---- level 1 ----
    k_fps<<<dim3(2), dim3(256), 0, stream>>>(xyz0, idx1, 512, 128, 512*3, 128);
    k_gather3<<<blk(2*128), THR, 0, stream>>>(xyz0, idx1, xyz1, 128, 512*3, 128, 128*3);
    k_ball_query<<<dim3(2, 2), dim3(64), 0, stream>>>(xyz1, xyz0, bqf, 128, 512, 16,
                                                      0.12f*0.12f, 384, 1536, 128*16);
    k_group_max<<<dim3((128*128 + 255)/256, 2), THR, 0, stream>>>(S0, bqf, feat1, 128, 128, 512, 16);
    if (t == 0) {
      k_fill<<<blk(65536), THR, 0, stream>>>(S1, 0.f, 65536);
      k_fill<<<blk(768), THR, 0, stream>>>(P1s, 0.f, 768);
    }
    k_ball_query<<<dim3(2, 2), dim3(64), 0, stream>>>(xyz1, P1s, bq1, 128, 128, 48,
                                                      0.24f*0.24f, 384, 384, 128*48);
    for (int b = 0; b < 2; ++b) {
      const int MNS = 128 * 48;
      k_build_corr<<<blk(MNS), THR, 0, stream>>>(S1 + b*32768, feat1 + b*16384, xyz1 + b*384,
                                                 P1s + b*384, bq1 + b*MNS, corrbf,
                                                 256, 128, 128, 48, 128, 416);
      gemm(stream, zW1p, corrbf, gbuf, nullptr, 256, MNS, 416, 416, 416, MNS, 0,0,0,1, 0,0);
      k_maxpool<<<blk(256*128), THR, 0, stream>>>(gbuf, zb1, Z1 + b*32768, 256, 128, 48, 1);
      gemm(stream, rW1p, corrbf, gbuf, nullptr, 256, MNS, 416, 416, 416, MNS, 0,0,0,1, 0,0);
      k_maxpool<<<blk(256*128), THR, 0, stream>>>(gbuf, rb1, R1 + b*32768, 256, 128, 48, 1);
      k_build_corr<<<blk(MNS), THR, 0, stream>>>(S1 + b*32768, nullptr, xyz1 + b*384,
                                                 P1s + b*384, bq1 + b*MNS, corrbf,
                                                 256, 0, 128, 48, 128, 288);
      gemm(stream, sW1p, corrbf, gbuf, nullptr, 256, MNS, 288, 288, 288, MNS, 0,0,0,1, 0,0);
      k_maxpool<<<blk(256*128), THR, 0, stream>>>(gbuf, sb1, So1 + b*32768, 256, 128, 48, 0);
    }
    k_fcin_pack<<<dim3((384*128 + 255)/256, 2), THR, 0, stream>>>(fcin1bf, feat1, R1, So1,
                                                                  128, 256, 128,
                                                                  49152, 16384, 32768);
    for (int b = 0; b < 2; ++b)
      gemm(stream, fW1p, fcin1bf + b*49152, Sn1 + b*32768, fb1, 256, 128, 384,
           384, 384, 128, 0,0,0,1, 1,2);
    k_gru_combine<<<blk(65536), THR, 0, stream>>>(S1, Z1, So1, Sn1, 65536);
    k_copy<<<blk(768), THR, 0, stream>>>(P1s, xyz1, 768);
    // append to sequence
    k_copy_b<<<blk(65536), THR, 0, stream>>>(featsAll + t*32768, S1, 32768, 131072, 32768, 2);
    k_copy_b<<<blk(768), THR, 0, stream>>>(possAll + t*384, xyz1, 384, 1536, 384, 2);
  }

  // ---- transformer ----
  for (int b = 0; b < 2; ++b) {
    packA(possAll + b*1536, posAbf + b*16384, 512, 3, 3, 32, 512);
    gemm(stream, posAbf + b*16384, posWp, xb + b*131072, posb, 512, 256, 32,
         32, 32, 256, 0,0,0,1, 2,0);
  }
  k_add<<<blk(262144), THR, 0, stream>>>(xb, featsAll, 262144);

  for (int l = 0; l < 2; ++l) {
    const float* ln1g = (const float*)d_in[19 + 11*l];
    const float* ln1b = (const float*)d_in[20 + 11*l];
    const float* outb = (const float*)d_in[23 + 11*l];
    const float* ln2g = (const float*)d_in[24 + 11*l];
    const float* ln2b = (const float*)d_in[25 + 11*l];
    const float* ff1b = (const float*)d_in[27 + 11*l];
    const float* ff2b = (const float*)d_in[29 + 11*l];

    k_layernorm<<<dim3(1024), dim3(256), 0, stream>>>(xb, ln1g, ln1b, hbf, 256);
    for (int b = 0; b < 2; ++b)
      gemm(stream, hbf + b*131072, qkvWp[l], qkv + b*393216, nullptr, 512, 768, 256,
           256, 256, 768, 0,0,0,1, 0,0);
    k_castbf<<<blk(786432), THR, 0, stream>>>(qkvbf, qkv, 786432);
    for (int b = 0; b < 2; ++b)   // scores per head: Q @ K^T
      gemm(stream, qkvbf + b*393216, qkvbf + b*393216 + 256, scores + (long long)b*2097152,
           nullptr, 512, 512, 32, 768, 768, 512, 32, 32, 262144, 8, 0, 0);
    k_softmax<<<dim3(8192), dim3(256), 0, stream>>>(scores, scorebf, 512, 0.17677669529663687f);
    for (int b = 0; b < 2; ++b)
      for (int h = 0; h < 8; ++h)  // V^T pack: Bt[d][token]
        packBt(qkv + b*393216 + 512 + h*32, vtbf + ((long)b*8 + h)*16384,
               32, 512, 768, 512, 0, 32);
    for (int b = 0; b < 2; ++b)   // attn @ V
      gemm(stream, scorebf + (long long)b*2097152, vtbf + b*131072, ob + b*131072,
           nullptr, 512, 32, 512, 512, 512, 256, 262144, 16384, 32, 8, 0, 0);
    k_castbf<<<blk(262144), THR, 0, stream>>>(obbf, ob, 262144);
    for (int b = 0; b < 2; ++b)
      gemm(stream, obbf + b*131072, outWp[l], hb + b*131072, outb, 512, 256, 256,
           256, 256, 256, 0,0,0,1, 2,0);
    k_add<<<blk(262144), THR, 0, stream>>>(xb, hb, 262144);
    k_layernorm<<<dim3(1024), dim3(256), 0, stream>>>(xb, ln2g, ln2b, hbf, 256);
    for (int b = 0; b < 2; ++b)
      gemm(stream, hbf + b*131072, ff1Wp[l], ffb + b*262144, ff1b, 512, 512, 256,
           256, 256, 512, 0,0,0,1, 2,4);
    k_castbf<<<blk(524288), THR, 0, stream>>>(ffbbf, ffb, 524288);
    for (int b = 0; b < 2; ++b)
      gemm(stream, ffbbf + b*262144, ff2Wp[l], hb + b*131072, ff2b, 512, 256, 512,
           512, 512, 256, 0,0,0,1, 2,0);
    k_add<<<blk(262144), THR, 0, stream>>>(xb, hb, 262144);
  }

  // ---- head ----
  k_relu_maxtok<<<dim3(1, 2), dim3(256), 0, stream>>>(xb, embmax, 512, 256);
  k_layernorm<<<dim3(2), dim3(256), 0, stream>>>(embmax, hlng, hlnb, hlnbf, 256);
  gemm(stream, hlnbf, l1Wp, l1o, l1b, 2, 512, 256, 256, 256, 512, 0,0,0,1, 2,0);
  k_castbf<<<blk(1024), THR, 0, stream>>>(l1obf, l1o, 1024);
  gemm(stream, l1obf, l2Wp, (float*)d_out, l2b, 2, 512, 512, 512, 512, 512, 0,0,0,1, 2,0);
}
